// SwinTransformerBlock_43267500540183
// MI455X (gfx1250) — matrix operand order repaired and checked by reading, not compiled, not run on hardware
//
#include <hip/hip_runtime.h>
#include <hip/hip_bf16.h>
#include <math.h>

// ---------------- CDNA5 WMMA types ----------------
typedef __attribute__((ext_vector_type(16))) _Float16 v16h;
typedef __attribute__((ext_vector_type(8)))  float    v8f;

union FragU { v16h v; uint4 u[2]; };

// A fragment (16x32 f16, M x K): lane m = l%16; lane-hi picks K {16..31} (contiguous 16 halves).
__device__ __forceinline__ v16h load_fragA(const _Float16* p) {
  FragU f;
  f.u[0] = *(const uint4*)(p);
  f.u[1] = *(const uint4*)(p + 8);
  return f.v;
}
// B fragment (32x16 f16, K x N) from N-major LDS tile: lane n = l%16;
// lanes 0-15 hold K=0..15, lanes 16-31 hold K=16..31 (contiguous 16 halves).
__device__ __forceinline__ v16h load_fragB(const _Float16* p) {
  FragU f;
  f.u[0] = *(const uint4*)(p);
  f.u[1] = *(const uint4*)(p + 8);
  return f.v;
}

// CDNA5 async copy: one b128 per lane, global -> LDS, tracked by ASYNCcnt.
__device__ __forceinline__ void async_b128(unsigned lds_off, const _Float16* gsrc) {
  asm volatile("global_load_async_to_lds_b128 %0, %1, off"
               :: "v"(lds_off), "v"((unsigned long long)(uintptr_t)gsrc)
               : "memory");
}
__device__ __forceinline__ void wait_async0() {
  asm volatile("s_wait_asynccnt 0x0" ::: "memory");
}

// ---------------- problem constants ----------------
constexpr int BATCH    = 32;
constexpr int HW       = 56;            // 56x56 tokens
constexpr int LSEQ     = HW * HW;       // 3136
constexpr int DIM      = 384;
constexpr int HEADS    = 12;
constexpr int HDIM     = 32;
constexpr int WIN      = 7;
constexpr int NTOK     = WIN * WIN;     // 49
constexpr int WTOT     = BATCH * 64;    // 2048 windows
constexpr int MROWS    = WTOT * NTOK;   // 100352 rows (div by 256)
constexpr int QKVN     = 3 * DIM;       // 1152
constexpr int MLPH     = 4 * DIM;       // 1536
constexpr float ATT_SCALE = 0.17677669529663687f;   // 32^-0.5

// window-major row -> token-major row (handles shift roll and window partition)
__device__ __forceinline__ int win2tok(int R) {
  int widx = R / NTOK, t = R - widx * NTOK;
  int b = widx >> 6, wrem = widx & 63;
  int wh = wrem >> 3, ww = wrem & 7;
  int i = t / WIN, j = t - i * WIN;
  int h = wh * WIN + i + 3; if (h >= HW) h -= HW;
  int w = ww * WIN + j + 3; if (w >= HW) w -= HW;
  return b * LSEQ + h * HW + w;
}

// ---------------- K0: weight convert fp32 [K x N] -> f16 N-major [N x K] ----
__global__ void conv_w_kernel(const float* __restrict__ W, _Float16* __restrict__ O,
                              int K, int N) {
  int o = blockIdx.x * 256 + threadIdx.x;
  if (o >= K * N) return;
  int n = o / K, k = o - n * K;
  O[o] = (_Float16)W[(size_t)k * N + n];
}

// ---------------- K1/K5: LayerNorm (one wave per token) ----------------
__global__ __launch_bounds__(256)
void ln_kernel(const float* __restrict__ X, const float* __restrict__ g,
               const float* __restrict__ b, _Float16* __restrict__ out,
               int gatherWin) {
  int wave = threadIdx.x >> 5, lane = threadIdx.x & 31;
  int R = blockIdx.x * 8 + wave;
  if (R >= MROWS) return;
  int src = gatherWin ? win2tok(R) : R;
  const float* xr = X + (size_t)src * DIM;
  float vals[12];
  float s = 0.f, s2 = 0.f;
#pragma unroll
  for (int i = 0; i < 12; ++i) {
    float xv = xr[i * 32 + lane];
    vals[i] = xv; s += xv; s2 += xv * xv;
  }
#pragma unroll
  for (int off = 16; off; off >>= 1) {
    s  += __shfl_xor(s,  off, 32);
    s2 += __shfl_xor(s2, off, 32);
  }
  float mean = s * (1.f / DIM);
  float var  = s2 * (1.f / DIM) - mean * mean;
  float rs   = rsqrtf(var + 1e-5f);
  _Float16* orow = out + (size_t)R * DIM;
#pragma unroll
  for (int i = 0; i < 12; ++i) {
    int c = i * 32 + lane;
    orow[c] = (_Float16)((vals[i] - mean) * rs * g[c] + b[c]);
  }
}

// ---------------- tiled WMMA GEMM, async double-buffered ------------------
// Block tile 256(M) x 64(N), 8 waves, each wave 32x64 = 2x4 WMMA tiles, K step 32.
// A: [MROWS x K] f16 row-major.  Bw: [N x K] f16 N-major (pre-transposed).
// MODE 0: out f16 = acc+bias                 (QKV)
// MODE 1: out f32[win2tok] = resid+acc+bias  (proj + window reverse + residual)
// MODE 2: out f16 = gelu(acc+bias)           (MLP fc1)
// MODE 3: out f32 = resid+acc+bias           (MLP fc2 + residual)
template<int MODE>
__global__ __launch_bounds__(256)
void gemm_wmma(const _Float16* __restrict__ A, const _Float16* __restrict__ Bw,
               const float* __restrict__ bias, const float* __restrict__ resid,
               float* __restrict__ outF, _Float16* __restrict__ outH,
               int N, int K) {
  __shared__ _Float16 sA[2][256 * 40];
  __shared__ _Float16 sB[2][64 * 40];
  const int tid  = threadIdx.x;
  const int lane = tid & 31, wave = tid >> 5;
  const int m0 = blockIdx.x * 256, n0 = blockIdx.y * 64;
  const int khiA = (lane >> 4) * 16, khiB = (lane >> 4) * 16, l15 = lane & 15;

  // async-stage one K-slab into LDS buffer `buf`
  auto stage = [&](int k0, int buf) {
#pragma unroll
    for (int i = 0; i < 4; ++i) {               // A: 256x32 halves = 1024 b128
      int idx = tid + i * 256;
      int row = idx >> 2, ko = (idx & 3) * 8;
      async_b128((unsigned)(uintptr_t)&sA[buf][row * 40 + ko],
                 &A[(size_t)(m0 + row) * K + k0 + ko]);
    }
    {                                           // B: 64x32 halves = 256 b128
      int row = tid >> 2, ko = (tid & 3) * 8;
      async_b128((unsigned)(uintptr_t)&sB[buf][row * 40 + ko],
                 &Bw[(size_t)(n0 + row) * K + k0 + ko]);
    }
  };

  v8f acc[2][4] = {};

  stage(0, 0);
  int buf = 0;
  for (int k0 = 0; k0 < K; k0 += 32, buf ^= 1) {
    wait_async0();          // this wave's loads for `buf` are in LDS
    __syncthreads();        // everyone's are; also: all waves done reading buf^1
    if (k0 + 32 < K) stage(k0 + 32, buf ^ 1);   // overlap next slab with compute

    v16h af[2], bf[4];
#pragma unroll
    for (int t = 0; t < 2; ++t)
      af[t] = load_fragA(&sA[buf][(wave * 32 + t * 16 + l15) * 40 + khiA]);
#pragma unroll
    for (int t = 0; t < 4; ++t)
      bf[t] = load_fragB(&sB[buf][(t * 16 + l15) * 40 + khiB]);
#pragma unroll
    for (int tm = 0; tm < 2; ++tm)
#pragma unroll
      for (int tn = 0; tn < 4; ++tn)
        acc[tm][tn] = __builtin_amdgcn_wmma_f32_16x16x32_f16(
            false, af[tm], false, bf[tn], (short)0, acc[tm][tn], false, false);
  }

  // epilogue: D layout -> m = v + 8*(lane>=16), n = lane%16 within tile
#pragma unroll
  for (int tm = 0; tm < 2; ++tm)
#pragma unroll
    for (int tn = 0; tn < 4; ++tn)
#pragma unroll
      for (int v = 0; v < 8; ++v) {
        int m = m0 + wave * 32 + tm * 16 + (lane >> 4) * 8 + v;
        int n = n0 + tn * 16 + l15;
        float val = acc[tm][tn][v] + bias[n];
        if constexpr (MODE == 0) {
          outH[(size_t)m * 1152 + n] = (_Float16)val;
        } else if constexpr (MODE == 2) {
          val = 0.5f * val * (1.0f + erff(val * 0.70710678118654752f));
          outH[(size_t)m * 1536 + n] = (_Float16)val;
        } else if constexpr (MODE == 1) {
          int dst = win2tok(m);
          outF[(size_t)dst * DIM + n] = resid[(size_t)dst * DIM + n] + val;
        } else {
          outF[(size_t)m * DIM + n] = resid[(size_t)m * DIM + n] + val;
        }
      }
}

// ---------------- K3: windowed attention (one block per (window, head)) ----
__global__ __launch_bounds__(128)
void attn_kernel(const _Float16* __restrict__ qkv,   // [MROWS x 1152]
                 const float* __restrict__ relT,     // [169 x 12]
                 _Float16* __restrict__ outC) {      // [MROWS x 384]
  __shared__ _Float16 sq[64 * 40];
  __shared__ _Float16 sk[64 * 40];
  __shared__ _Float16 sv[32 * 72];   // [d][token]  (B operand for P@V)
  __shared__ float    sS[64 * 68];
  __shared__ _Float16 sP[64 * 72];

  const int widx = blockIdx.x, head = blockIdx.y;
  const int tid = threadIdx.x;
  const size_t rowBase = (size_t)widx * NTOK;

  // load q, k tiles (pad rows 49..63 with zeros)
  for (int idx = tid; idx < 256; idx += 128) {
    int t = idx >> 2, dg = (idx & 3) * 8;
    uint4 zq = {0, 0, 0, 0}, zk = zq;
    if (t < NTOK) {
      const _Float16* row = qkv + (rowBase + t) * QKVN + head * HDIM + dg;
      zq = *(const uint4*)(row);
      zk = *(const uint4*)(row + DIM);
    }
    *(uint4*)&sq[t * 40 + dg] = zq;
    *(uint4*)&sk[t * 40 + dg] = zk;
  }
  // load v transposed: sv[d][token]
  for (int idx = tid; idx < 2048; idx += 128) {
    int t = idx >> 5, d = idx & 31;
    _Float16 val = (_Float16)0.0f;
    if (t < NTOK) val = qkv[(rowBase + t) * QKVN + 2 * DIM + head * HDIM + d];
    sv[d * 72 + t] = val;
  }
  __syncthreads();

  const int lane = tid & 31, wave = tid >> 5;
  const int l15 = lane & 15, khiA = (lane >> 4) * 16, khiB = (lane >> 4) * 16;

  // S = q @ k^T : wave handles one 16-row strip, 4 N-tiles, K=32 single step
  {
    v16h aq = load_fragA(&sq[(wave * 16 + l15) * 40 + khiA]);
#pragma unroll
    for (int nt = 0; nt < 4; ++nt) {
      v16h bk = load_fragB(&sk[(nt * 16 + l15) * 40 + khiB]);
      v8f c = {};
      c = __builtin_amdgcn_wmma_f32_16x16x32_f16(false, aq, false, bk,
                                                 (short)0, c, false, false);
#pragma unroll
      for (int v = 0; v < 8; ++v)
        sS[(wave * 16 + (lane >> 4) * 8 + v) * 68 + nt * 16 + l15] = c[v];
    }
  }
  __syncthreads();

  // softmax rows (scale + relative position bias computed on the fly)
  if (tid < 64) {
    int r = tid;
    int rq = r < NTOK ? r : NTOK - 1;
    int ni = rq / WIN, nj = rq - ni * WIN;
    float mx = -1e30f;
    for (int m = 0; m < NTOK; ++m) {
      int mi = m / WIN, mj = m - mi * WIN;
      int idx = (ni - mi + WIN - 1) * (2 * WIN - 1) + (nj - mj + WIN - 1);
      float val = sS[r * 68 + m] * ATT_SCALE + relT[idx * HEADS + head];
      sS[r * 68 + m] = val;
      mx = fmaxf(mx, val);
    }
    float sum = 0.f;
    for (int m = 0; m < NTOK; ++m) {
      float e = __expf(sS[r * 68 + m] - mx);
      sS[r * 68 + m] = e;
      sum += e;
    }
    float inv = 1.0f / sum;
    for (int m = 0; m < NTOK; ++m) sP[r * 72 + m] = (_Float16)(sS[r * 68 + m] * inv);
    for (int m = NTOK; m < 64; ++m) sP[r * 72 + m] = (_Float16)0.0f;
  }
  __syncthreads();

  // O = P @ V : M tile per wave, N = 32 (2 tiles), K = 64 (2 steps)
  v8f o[2] = {};
#pragma unroll
  for (int ks = 0; ks < 2; ++ks) {
    v16h ap = load_fragA(&sP[(wave * 16 + l15) * 72 + ks * 32 + khiA]);
#pragma unroll
    for (int nt = 0; nt < 2; ++nt) {
      v16h bv = load_fragB(&sv[(nt * 16 + l15) * 72 + ks * 32 + khiB]);
      o[nt] = __builtin_amdgcn_wmma_f32_16x16x32_f16(false, ap, false, bv,
                                                     (short)0, o[nt], false, false);
    }
  }
#pragma unroll
  for (int nt = 0; nt < 2; ++nt)
#pragma unroll
    for (int v = 0; v < 8; ++v) {
      int t = wave * 16 + (lane >> 4) * 8 + v;
      if (t < NTOK)
        outC[(rowBase + t) * DIM + head * HDIM + nt * 16 + l15] = (_Float16)o[nt][v];
    }
}

// ---------------- host launch ----------------
extern "C" void kernel_launch(void* const* d_in, const int* in_sizes, int n_in,
                              void* d_out, int out_size, void* d_ws, size_t ws_size,
                              hipStream_t stream) {
  const float* x      = (const float*)d_in[0];
  const float* n1g    = (const float*)d_in[1];
  const float* n1b    = (const float*)d_in[2];
  const float* qkv_w  = (const float*)d_in[3];
  const float* qkv_b  = (const float*)d_in[4];
  const float* relT   = (const float*)d_in[5];
  const float* proj_w = (const float*)d_in[6];
  const float* proj_b = (const float*)d_in[7];
  const float* n2g    = (const float*)d_in[8];
  const float* n2b    = (const float*)d_in[9];
  const float* m1w    = (const float*)d_in[10];
  const float* m1b    = (const float*)d_in[11];
  const float* m2w    = (const float*)d_in[12];
  const float* m2b    = (const float*)d_in[13];
  float* out = (float*)d_out;

  // workspace layout (bytes)
  const size_t off_wqkv = 0;                                   // 1152*384 f16
  const size_t off_wprj = off_wqkv + (size_t)QKVN * DIM * 2;
  const size_t off_wm1  = off_wprj + (size_t)DIM * DIM * 2;
  const size_t off_wm2  = off_wm1  + (size_t)MLPH * DIM * 2;
  const size_t off_A    = off_wm2  + (size_t)MLPH * DIM * 2;   // [MROWS x 384] f16
  const size_t off_B    = off_A + (size_t)MROWS * DIM * 2;     // [MROWS x 1536] f16
  const size_t off_C    = off_B + (size_t)MROWS * MLPH * 2;    // [MROWS x 384] f16
  const size_t off_X    = off_C + (size_t)MROWS * DIM * 2;     // [MROWS x 384] f32
  const size_t need     = off_X + (size_t)MROWS * DIM * 4;
  if (ws_size < need) return;

  char* ws = (char*)d_ws;
  _Float16* w16_qkv = (_Float16*)(ws + off_wqkv);
  _Float16* w16_prj = (_Float16*)(ws + off_wprj);
  _Float16* w16_m1  = (_Float16*)(ws + off_wm1);
  _Float16* w16_m2  = (_Float16*)(ws + off_wm2);
  _Float16* bufA    = (_Float16*)(ws + off_A);
  _Float16* bufB    = (_Float16*)(ws + off_B);
  _Float16* bufC    = (_Float16*)(ws + off_C);
  float*    bufX    = (float*)   (ws + off_X);

  // K0: weight conversion (transposed to N-major f16)
  conv_w_kernel<<<(QKVN * DIM + 255) / 256, 256, 0, stream>>>(qkv_w, w16_qkv, DIM, QKVN);
  conv_w_kernel<<<(DIM * DIM + 255) / 256, 256, 0, stream>>>(proj_w, w16_prj, DIM, DIM);
  conv_w_kernel<<<(DIM * MLPH + 255) / 256, 256, 0, stream>>>(m1w, w16_m1, DIM, MLPH);
  conv_w_kernel<<<(MLPH * DIM + 255) / 256, 256, 0, stream>>>(m2w, w16_m2, MLPH, DIM);

  // K1: LN1 + shift + window partition  -> bufA (window-major f16)
  ln_kernel<<<MROWS / 8, 256, 0, stream>>>(x, n1g, n1b, bufA, 1);

  // K2: QKV GEMM  [MROWS x 384] @ [384 x 1152] -> bufB f16
  gemm_wmma<0><<<dim3(MROWS / 256, QKVN / 64), 256, 0, stream>>>(
      bufA, w16_qkv, qkv_b, nullptr, nullptr, bufB, QKVN, DIM);

  // K3: attention per (window, head) -> bufC f16
  attn_kernel<<<dim3(WTOT, HEADS), 128, 0, stream>>>(bufB, relT, bufC);

  // K4: proj GEMM + window reverse + residual -> bufX f32 (token-major)
  gemm_wmma<1><<<dim3(MROWS / 256, DIM / 64), 256, 0, stream>>>(
      bufC, w16_prj, proj_b, x, bufX, nullptr, DIM, DIM);

  // K5: LN2 -> bufA (token-major f16)
  ln_kernel<<<MROWS / 8, 256, 0, stream>>>(bufX, n2g, n2b, bufA, 0);

  // K6: MLP fc1 + GELU -> bufB f16
  gemm_wmma<2><<<dim3(MROWS / 256, MLPH / 64), 256, 0, stream>>>(
      bufA, w16_m1, m1b, nullptr, nullptr, bufB, MLPH, DIM);

  // K7: MLP fc2 + residual -> d_out f32
  gemm_wmma<3><<<dim3(MROWS / 256, DIM / 64), 256, 0, stream>>>(
      bufB, w16_m2, m2b, bufX, out, nullptr, DIM, MLPH);
}
